// DGN1_70428873720408
// MI455X (gfx1250) — matrix operand order repaired and checked by reading, not compiled, not run on hardware
//
#include <hip/hip_runtime.h>
#include <hip/hip_bf16.h>
#include <math.h>

// Problem constants (match reference)
#define B_ 4
#define T_ 4096
#define D_ 1024
#define K_ 8
#define QT_ (T_ / 16)          // 256 query tiles per batch
#define NWAVE 8                // waves per top-k workgroup
#define SENT (-3.0e38f)

typedef __attribute__((ext_vector_type(16))) _Float16 v16h;
typedef __attribute__((ext_vector_type(8)))  _Float16 v8h;
typedef __attribute__((ext_vector_type(4)))  _Float16 v4h;
typedef __attribute__((ext_vector_type(8)))  float    v8f;
typedef __attribute__((ext_vector_type(4)))  float    v4f;

// ---------------------------------------------------------------------------
// Pass 1: row L2-normalize x (fp32) -> xn (f16), xn = x / (||x|| + 1e-8)
// One block (256 threads) per row; each thread owns 4 contiguous features.
// ---------------------------------------------------------------------------
__global__ __launch_bounds__(256) void dgn_normalize(const float* __restrict__ x,
                                                     _Float16* __restrict__ xn) {
    const int row = blockIdx.x;                 // 0 .. B*T-1
    const int tid = threadIdx.x;                // 0 .. 255
    const float* xr = x + (size_t)row * D_;

    v4f v = ((const v4f*)xr)[tid];
    float ss = v.x * v.x + v.y * v.y + v.z * v.z + v.w * v.w;

    __shared__ float red[256];
    red[tid] = ss;
    __syncthreads();
    for (int off = 128; off > 0; off >>= 1) {
        if (tid < off) red[tid] += red[tid + off];
        __syncthreads();
    }
    const float inv = 1.0f / (sqrtf(red[0]) + 1e-8f);

    v4h h;
    h.x = (_Float16)(v.x * inv);
    h.y = (_Float16)(v.y * inv);
    h.z = (_Float16)(v.z * inv);
    h.w = (_Float16)(v.w * inv);
    ((v4h*)(xn + (size_t)row * D_))[tid] = h;
}

// ---------------------------------------------------------------------------
// Pass 2: causal top-8 cosine neighbors via WMMA f16 -> f32.
// grid = B * QT_ workgroups, 256 threads (8 waves of 32).
//   - 16x1024 f16 query tile staged in LDS (32 KB).
//   - wave w handles j-tiles w, w+8, ...  (disjoint -> mergeable top-8 sets)
//   - per j-tile: 32 x v_wmma_f32_16x16x32_f16 accumulate a 16x16 sim tile
//   - lanes 0..15 scan the tile (causal mask) maintaining a register top-8
//   - block-wide merge of 8 wave-local top-8 lists -> global top-8 indices
// ---------------------------------------------------------------------------
__global__ __launch_bounds__(256) void dgn_topk(const _Float16* __restrict__ xn,
                                                int* __restrict__ topk_idx) {
    __shared__ __align__(16) _Float16 Atile[16 * D_];      // 32 KB
    __shared__ float simbuf[NWAVE][16][16];                // 8 KB
    __shared__ float mval[NWAVE][16][K_];                  // 4 KB
    __shared__ int   midx[NWAVE][16][K_];                  // 4 KB

    const int wg   = blockIdx.x;
    const int b    = wg / QT_;
    const int qt   = wg % QT_;
    const int tid  = threadIdx.x;
    const int wave = tid >> 5;
    const int lane = tid & 31;

    const _Float16* xb = xn + (size_t)b * T_ * D_;

    // cooperative load of the 16-row query tile into LDS (2048 x 16B chunks)
    {
        const v8h* src = (const v8h*)(xb + (size_t)qt * 16 * D_);
        v8h* dst = (v8h*)Atile;
        for (int i = tid; i < 16 * D_ / 8; i += 256) dst[i] = src[i];
    }
    __syncthreads();

    // register-resident top-8 (values + indices), fully unrolled accesses only
    float tv0 = SENT, tv1 = SENT, tv2 = SENT, tv3 = SENT,
          tv4 = SENT, tv5 = SENT, tv6 = SENT, tv7 = SENT;
    int   ti0 = -1, ti1 = -1, ti2 = -1, ti3 = -1,
          ti4 = -1, ti5 = -1, ti6 = -1, ti7 = -1;
    float tmin = SENT;

    const int i_global = qt * 16 + (lane & 15);   // query row owned at scan time
    const int m0 = (lane & 16) ? 8 : 0;           // C-fragment row offset
    const int nn = lane & 15;                     // C-fragment column

    // per-lane source offsets for the WMMA fragments
    const int a_off = (lane & 15) * D_ + ((lane & 16) ? 8 : 0);   // interleaved A
    const int b_off = (lane & 15) * D_ + ((lane & 16) ? 8 : 0);   // interleaved B

    for (int jt = wave; jt <= qt; jt += NWAVE) {
        const _Float16* bb = xb + (size_t)jt * 16 * D_;

        v8f acc = {};
        for (int kc = 0; kc < D_; kc += 32) {
            // A fragment (16x32 f16) from LDS: halves 0-7 = K+0..7 (or 8..15),
            // halves 8-15 = K+16..23 (or 24..31) per the CDNA5 A layout.
            const _Float16* ap = Atile + a_off + kc;
            v8h alo = *(const v8h*)ap;
            v8h ahi = *(const v8h*)(ap + 16);
            v16h afrag = __builtin_shufflevector(alo, ahi,
                0, 1, 2, 3, 4, 5, 6, 7, 8, 9, 10, 11, 12, 13, 14, 15);

            // B fragment (32x16 f16) from global: same interleaved K layout as A
            // (halves 0-7 = K+0..7 (or 8..15), halves 8-15 = K+16..23 (or 24..31)).
            const _Float16* bp = bb + b_off + kc;
            v8h blo = *(const v8h*)bp;
            v8h bhi = *(const v8h*)(bp + 16);
            v16h bfrag = __builtin_shufflevector(blo, bhi,
                0, 1, 2, 3, 4, 5, 6, 7, 8, 9, 10, 11, 12, 13, 14, 15);

            acc = __builtin_amdgcn_wmma_f32_16x16x32_f16(
                false, afrag, false, bfrag, (short)0, acc, false, false);
        }

        // stage C tile to this wave's LDS buffer (cross-lane row gather next)
        #pragma unroll
        for (int r = 0; r < 8; ++r) simbuf[wave][m0 + r][nn] = acc[r];
        asm volatile("s_wait_dscnt 0" ::: "memory");   // wave-internal LDS RAW

        if (lane < 16) {
            const int m = lane;
            #pragma unroll
            for (int n2 = 0; n2 < 16; ++n2) {
                const int j = jt * 16 + n2;
                const float v = simbuf[wave][m][n2];
                if (j <= i_global && v > tmin) {
                    // replace the (first) slot holding the current minimum
                    bool done = false;
                    if (!done && tv0 == tmin) { tv0 = v; ti0 = j; done = true; }
                    if (!done && tv1 == tmin) { tv1 = v; ti1 = j; done = true; }
                    if (!done && tv2 == tmin) { tv2 = v; ti2 = j; done = true; }
                    if (!done && tv3 == tmin) { tv3 = v; ti3 = j; done = true; }
                    if (!done && tv4 == tmin) { tv4 = v; ti4 = j; done = true; }
                    if (!done && tv5 == tmin) { tv5 = v; ti5 = j; done = true; }
                    if (!done && tv6 == tmin) { tv6 = v; ti6 = j; done = true; }
                    if (!done && tv7 == tmin) { tv7 = v; ti7 = j; done = true; }
                    tmin = fminf(fminf(fminf(tv0, tv1), fminf(tv2, tv3)),
                                 fminf(fminf(tv4, tv5), fminf(tv6, tv7)));
                }
            }
        }
        asm volatile("s_wait_dscnt 0" ::: "memory");   // reads done before reuse
    }

    // publish per-wave top-8 lists
    if (lane < 16) {
        const int m = lane;
        mval[wave][m][0] = tv0; midx[wave][m][0] = ti0;
        mval[wave][m][1] = tv1; midx[wave][m][1] = ti1;
        mval[wave][m][2] = tv2; midx[wave][m][2] = ti2;
        mval[wave][m][3] = tv3; midx[wave][m][3] = ti3;
        mval[wave][m][4] = tv4; midx[wave][m][4] = ti4;
        mval[wave][m][5] = tv5; midx[wave][m][5] = ti5;
        mval[wave][m][6] = tv6; midx[wave][m][6] = ti6;
        mval[wave][m][7] = tv7; midx[wave][m][7] = ti7;
    }
    __syncthreads();

    // merge 8 disjoint top-8 lists -> global top-8 (one lane per query row)
    if (tid < 16) {
        const int m = tid;
        const int ig = qt * 16 + m;
        int* outp = topk_idx + ((size_t)b * T_ + ig) * K_;
        for (int s = 0; s < K_; ++s) {
            float best = SENT;
            int bw = 0, bs = 0;
            for (int w = 0; w < NWAVE; ++w) {
                for (int t = 0; t < K_; ++t) {
                    const float v = mval[w][m][t];
                    if (v > best) { best = v; bw = w; bs = t; }
                }
            }
            outp[s] = (best > -1.0e38f) ? midx[bw][m][bs] : -1;
            mval[bw][m][bs] = SENT;
        }
    }
}

// ---------------------------------------------------------------------------
// Pass 3: gather top-K neighbors, mean-aggregate, gated GELU epilogue.
// One block (256 threads) per row; each thread owns 4 contiguous features.
// ---------------------------------------------------------------------------
__global__ __launch_bounds__(256) void dgn_epilogue(const float* __restrict__ x,
                                                    const float* __restrict__ gain,
                                                    const float* __restrict__ bias,
                                                    const float* __restrict__ log_mix,
                                                    const float* __restrict__ log_scale,
                                                    const int* __restrict__ topk_idx,
                                                    float* __restrict__ out) {
    const int row = blockIdx.x;              // 0 .. B*T-1
    const int b   = row / T_;
    const int i   = row % T_;
    const int tid = threadIdx.x;

    const float mix = 1.0f / (1.0f + expf(-log_mix[0]));
    const float lz  = log_scale[0];
    const float scale = ((lz > 20.0f) ? lz : log1pf(expf(lz))) + 0.01f;

    const int* idxp = topk_idx + (size_t)row * K_;
    int idxs[K_];
    #pragma unroll
    for (int s = 0; s < K_; ++s) idxs[s] = idxp[s];
    const int cvalid = (i + 1 < K_) ? (i + 1) : K_;
    const float rcnt = 1.0f / (float)(cvalid > 0 ? cvalid : 1);

    const float* xb = x + (size_t)b * T_ * D_;
    v4f acc = {};
    #pragma unroll
    for (int s = 0; s < K_; ++s) {
        const int j = idxs[s];
        if (j >= 0) {
            v4f g = ((const v4f*)(xb + (size_t)j * D_))[tid];
            acc.x += g.x; acc.y += g.y; acc.z += g.z; acc.w += g.w;
        }
    }

    v4f xv = ((const v4f*)(xb + (size_t)i * D_))[tid];
    v4f gv = ((const v4f*)gain)[tid];
    v4f bv = ((const v4f*)bias)[tid];

    const float omix = 1.0f - mix;
    v4f r;
    {
        float m0 = acc.x * rcnt, m1 = acc.y * rcnt, m2 = acc.z * rcnt, m3 = acc.w * rcnt;
        float z0 = (mix * xv.x + omix * m0) * gv.x + bv.x;
        float z1 = (mix * xv.y + omix * m1) * gv.y + bv.y;
        float z2 = (mix * xv.z + omix * m2) * gv.z + bv.z;
        float z3 = (mix * xv.w + omix * m3) * gv.w + bv.w;
        const float c = 0.70710678118654752f;
        r.x = 0.5f * z0 * (1.0f + erff(z0 * c)) * scale;
        r.y = 0.5f * z1 * (1.0f + erff(z1 * c)) * scale;
        r.z = 0.5f * z2 * (1.0f + erff(z2 * c)) * scale;
        r.w = 0.5f * z3 * (1.0f + erff(z3 * c)) * scale;
    }
    ((v4f*)(out + (size_t)row * D_))[tid] = r;
}

// ---------------------------------------------------------------------------
extern "C" void kernel_launch(void* const* d_in, const int* in_sizes, int n_in,
                              void* d_out, int out_size, void* d_ws, size_t ws_size,
                              hipStream_t stream) {
    (void)in_sizes; (void)n_in; (void)out_size; (void)ws_size;

    const float* x         = (const float*)d_in[0];
    const float* gain      = (const float*)d_in[1];
    const float* bias      = (const float*)d_in[2];
    const float* log_mix   = (const float*)d_in[3];
    const float* log_scale = (const float*)d_in[4];
    float* out = (float*)d_out;

    // workspace layout: [ xn f16 : B*T*D*2 = 32 MB ][ topk idx : B*T*8*4 = 512 KB ]
    _Float16* xn = (_Float16*)d_ws;
    int* topk = (int*)((char*)d_ws + (size_t)B_ * T_ * D_ * sizeof(_Float16));

    dgn_normalize<<<B_ * T_, 256, 0, stream>>>(x, xn);
    dgn_topk<<<B_ * QT_, 32 * NWAVE, 0, stream>>>(xn, topk);
    dgn_epilogue<<<B_ * T_, 256, 0, stream>>>(x, gain, bias, log_mix, log_scale,
                                              topk, out);
}
